// LinearRNNwithSoftExp_26706106646543
// MI455X (gfx1250) — matrix operand order repaired and checked by reading, not compiled, not run on hardware
//
#include <hip/hip_runtime.h>
#include <hip/hip_bf16.h>

// Problem constants (reference: T=1024, B=512, D=128, H=256)
#define T_STEPS 1024
#define BATCH   512
#define DIM     128
#define HID     256
#define HID2    512
#define BT      16    // batch rows per workgroup (independent recurrences)
#define NTHREADS 256  // 8 wave32 waves

typedef __attribute__((ext_vector_type(16))) __bf16 v16bf;
typedef __attribute__((ext_vector_type(8)))  __bf16 v8bf;
typedef __attribute__((ext_vector_type(8)))  float  v8f;

__device__ __forceinline__ unsigned short f2bf(float f) {
    union { float f; unsigned u; } v; v.f = f;
    unsigned u = v.u;
    u += 0x7FFFu + ((u >> 16) & 1u);   // round-to-nearest-even
    return (unsigned short)(u >> 16);
}

__device__ __forceinline__ float fast_rcp(float x) {
#if __has_builtin(__builtin_amdgcn_rcpf)
    return __builtin_amdgcn_rcpf(x);     // v_rcp_f32
#else
    return 1.0f / x;
#endif
}

__device__ __forceinline__ float fast_tanh(float z) {
#if __has_builtin(__builtin_amdgcn_tanhf)
    return __builtin_amdgcn_tanhf(z);    // gfx1250 v_tanh_f32
#else
    // tanh(z) = 1 - 2/(exp(2z)+1); exact at +-inf saturation
    float e = __expf(2.0f * z);
    return 1.0f - 2.0f * fast_rcp(e + 1.0f);
#endif
}

__device__ __forceinline__ v16bf cat16(v8bf lo, v8bf hi) {
    return __builtin_shufflevector(lo, hi, 0,1,2,3,4,5,6,7,8,9,10,11,12,13,14,15);
}

// B-fragment (32x16 bf16, K x N): lane l holds column N = l&15 == row (n0 + l&15)
// of row-major W[nrows][ldk]; halves K = k0 + (l>=16 ? 16 : 0) .. +15 (contiguous 32B).
__device__ __forceinline__ v16bf load_bfrag(const unsigned short* __restrict__ W,
                                            int ldk, int n0, int k0, int lane) {
    const unsigned short* p = W + (size_t)(n0 + (lane & 15)) * ldk + k0 + ((lane & 16) ? 16 : 0);
    v8bf lo = *(const v8bf*)p;
    v8bf hi = *(const v8bf*)(p + 8);
    return cat16(lo, hi);
}

// A-fragment (16x32 bf16, M x K) from LDS row-major [16][ldk]:
// lane l holds row M = l&15; halves K = c0..c0+15 (contiguous), c0 = k0 + (l>=16 ? 16 : 0).
__device__ __forceinline__ v16bf load_afrag(const unsigned short* A, int ldk, int k0, int lane) {
    const unsigned short* p = A + (lane & 15) * ldk + k0 + ((lane & 16) ? 16 : 0);
    v8bf lo = *(const v8bf*)p;
    v8bf hi = *(const v8bf*)(p + 8);
    return cat16(lo, hi);
}

// Prologue: quantize weights to bf16 once (L2-resident thereafter).
__global__ void convert_weights_kernel(const float* __restrict__ ih_w,
                                       const float* __restrict__ hh_w,
                                       unsigned short* __restrict__ hh_bf,
                                       unsigned short* __restrict__ ih_bf) {
    int i = blockIdx.x * blockDim.x + threadIdx.x;
    if (i < HID2 * HID) hh_bf[i] = f2bf(hh_w[i]);
    if (i < HID * DIM)  ih_bf[i] = f2bf(ih_w[i]);
}

__global__ void __launch_bounds__(NTHREADS)
rnn_recurrence(const float* __restrict__ data,     // [T,B,D] f32
               const float* __restrict__ ih_b,     // [H]
               const float* __restrict__ hh_b,     // [2H]
               const int*   __restrict__ lengths,  // [B]
               const unsigned short* __restrict__ hhw_bf, // [2H][H] bf16
               const unsigned short* __restrict__ ihw_bf, // [H][D]  bf16
               float* __restrict__ out)            // [B,H] f32
{
    __shared__ unsigned short s_hbf[BT * HID];  // hidden, bf16 (WMMA A operand)  8 KB
    __shared__ unsigned short s_xbf[BT * DIM];  // x_t tile, bf16                 4 KB

    const int tid  = threadIdx.x;
    const int lane = tid & 31;
    const int wv   = tid >> 5;           // wave id 0..7
    const int b0   = blockIdx.x * BT;    // this WG's batch rows

    // zero-init hidden state (bf16 copy; f32 carry lives in registers)
    for (int i = tid; i < BT * HID; i += NTHREADS) s_hbf[i] = 0;

    // Per-lane constants (C-layout: lane owns rows M = mbase + r, column col = j*16 + (lane&15))
    const int mbase = (lane & 16) ? 8 : 0;
    const int n     = lane & 15;
    int len8[8];
    #pragma unroll
    for (int r = 0; r < 8; ++r) len8[r] = lengths[b0 + mbase + r];

    // WG-uniform loop bound: hidden is frozen once t >= max(lengths in tile)
    int tmax = 0;
    #pragma unroll
    for (int r = 0; r < BT; ++r) tmax = max(tmax, lengths[b0 + r]);

    float biasX[2], biasA[2], biasH[2];
    #pragma unroll
    for (int jj = 0; jj < 2; ++jj) {
        const int col = (2 * wv + jj) * 16 + n;
        biasX[jj] = ih_b[col];
        biasA[jj] = hh_b[col];
        biasH[jj] = hh_b[col + HID];
    }

    // Preload ALL weight B-fragments into registers (t-invariant; ~320 VGPRs)
    v16bf b_x[2][4], b_a[2][8], b_h[2][8];
    #pragma unroll
    for (int jj = 0; jj < 2; ++jj) {
        const int j = 2 * wv + jj;
        #pragma unroll
        for (int k = 0; k < 4; ++k) b_x[jj][k] = load_bfrag(ihw_bf, DIM, j * 16, k * 32, lane);
        #pragma unroll
        for (int k = 0; k < 8; ++k) b_a[jj][k] = load_bfrag(hhw_bf, HID, j * 16, k * 32, lane);
        #pragma unroll
        for (int k = 0; k < 8; ++k) b_h[jj][k] = load_bfrag(hhw_bf, HID, j * 16 + HID, k * 32, lane);
    }

    // f32 hidden carry: this lane exclusively owns cells (mbase+r, 2wv*16+jj*16+n)
    float hcarry[2][8];
    #pragma unroll
    for (int jj = 0; jj < 2; ++jj)
        #pragma unroll
        for (int r = 0; r < 8; ++r) hcarry[jj][r] = 0.0f;

    const int xrow = tid >> 4;        // 0..15
    const int xcol = (tid & 15) * 8;  // 0..120, 8 f32 each

    __syncthreads();

    for (int t = 0; t < tmax; ++t) {
        // ---- stage x_t tile into LDS as bf16 (16 rows x 128) ----
        {
            const float* xp = data + ((size_t)t * BATCH + b0 + xrow) * DIM + xcol;
            float4 f0 = *(const float4*)xp;
            float4 f1 = *(const float4*)(xp + 4);
            uint4 u;
            u.x = (unsigned)f2bf(f0.x) | ((unsigned)f2bf(f0.y) << 16);
            u.y = (unsigned)f2bf(f0.z) | ((unsigned)f2bf(f0.w) << 16);
            u.z = (unsigned)f2bf(f1.x) | ((unsigned)f2bf(f1.y) << 16);
            u.w = (unsigned)f2bf(f1.z) | ((unsigned)f2bf(f1.w) << 16);
            *(uint4*)&s_xbf[xrow * DIM + xcol] = u;
            if (t + 1 < T_STEPS) { // pull next step's tile toward the WGP (global_prefetch_b8)
                __builtin_prefetch(data + ((size_t)(t + 1) * BATCH + b0 + xrow) * DIM + xcol, 0, 3);
            }
        }
        __syncthreads();   // x + hidden LDS ready for reads

        // ---- load A fragments (shared across this wave's output tiles) ----
        v16bf a_h[8];
        #pragma unroll
        for (int k = 0; k < 8; ++k) a_h[k] = load_afrag(s_hbf, HID, k * 32, lane);
        v16bf a_x[4];
        #pragma unroll
        for (int k = 0; k < 4; ++k) a_x[k] = load_afrag(s_xbf, DIM, k * 32, lane);

        __syncthreads();   // all reads of hidden done before anyone overwrites it

        #pragma unroll
        for (int jj = 0; jj < 2; ++jj) {
            const int j = 2 * wv + jj;   // output column tile 0..15
            v8f accx = {0.f,0.f,0.f,0.f,0.f,0.f,0.f,0.f};
            v8f acca = {0.f,0.f,0.f,0.f,0.f,0.f,0.f,0.f};
            v8f acch = {0.f,0.f,0.f,0.f,0.f,0.f,0.f,0.f};

            #pragma unroll
            for (int k = 0; k < 4; ++k)   // x @ ih_w^T  (K = 128)
                accx = __builtin_amdgcn_wmma_f32_16x16x32_bf16(
                           false, a_x[k], false, b_x[jj][k], (short)0, accx, false, false);
            #pragma unroll
            for (int k = 0; k < 8; ++k)   // hidden @ hh_w^T, alpha half (K = 256)
                acca = __builtin_amdgcn_wmma_f32_16x16x32_bf16(
                           false, a_h[k], false, b_a[jj][k], (short)0, acca, false, false);
            #pragma unroll
            for (int k = 0; k < 8; ++k)   // hidden @ hh_w^T, h half
                acch = __builtin_amdgcn_wmma_f32_16x16x32_bf16(
                           false, a_h[k], false, b_h[jj][k], (short)0, acch, false, false);

            // ---- elementwise gate + tanh + length masking, update hidden ----
            const int col = j * 16 + n;
            #pragma unroll
            for (int r = 0; r < 8; ++r) {
                float av    = acca[r] + biasA[jj];
                float alpha = fast_rcp(1.0f + __expf(-av));        // sigmoid
                alpha = fmaxf(alpha, 1e-20f);
                float hv    = acch[r] + biasH[jj];
                float soft  = (exp2f(alpha * hv) - 1.0f) * fast_rcp(alpha) + alpha;
                float nh    = fast_tanh(accx[r] + biasX[jj] + soft);
                float val   = (t < len8[r]) ? nh : hcarry[jj][r];
                hcarry[jj][r] = val;
                s_hbf[(mbase + r) * HID + col] = f2bf(val);
            }
        }
        // next-iteration top __syncthreads orders these writes before the next reads
    }

    // ---- write final hidden carry straight from registers to d_out [1,B,H] ----
    #pragma unroll
    for (int jj = 0; jj < 2; ++jj) {
        const int col = (2 * wv + jj) * 16 + n;
        #pragma unroll
        for (int r = 0; r < 8; ++r)
            out[(size_t)(b0 + mbase + r) * HID + col] = hcarry[jj][r];
    }
}

extern "C" void kernel_launch(void* const* d_in, const int* in_sizes, int n_in,
                              void* d_out, int out_size, void* d_ws, size_t ws_size,
                              hipStream_t stream) {
    const float* data    = (const float*)d_in[0];  // [T,B,D]
    const float* ih_w    = (const float*)d_in[1];  // [H,D]
    const float* ih_b    = (const float*)d_in[2];  // [H]
    const float* hh_w    = (const float*)d_in[3];  // [2H,H]
    const float* hh_b    = (const float*)d_in[4];  // [2H]
    const int*   lengths = (const int*)d_in[5];    // [B]

    unsigned short* hh_bf = (unsigned short*)d_ws;         // 2H*H bf16 = 256 KB
    unsigned short* ih_bf = hh_bf + (size_t)HID2 * HID;    // H*D  bf16 =  64 KB
    float* out = (float*)d_out;

    convert_weights_kernel<<<(HID2 * HID + NTHREADS - 1) / NTHREADS, NTHREADS, 0, stream>>>(
        ih_w, hh_w, hh_bf, ih_bf);
    rnn_recurrence<<<BATCH / BT, NTHREADS, 0, stream>>>(
        data, ih_b, hh_b, lengths, hh_bf, ih_bf, out);
}
